// MambaLayer_77515569758967
// MI455X (gfx1250) — matrix operand order repaired and checked by reading, not compiled, not run on hardware
//
#include <hip/hip_runtime.h>
#include <hip/hip_bf16.h>
#include <math.h>

// ---------------------------------------------------------------------------
// Mamba layer for MI455X (gfx1250, wave32).
// GEMMs: v_wmma_f32_16x16x32_bf16 with bf16 hi/lo compensation
// (A*B ~= Ah*Bh + Ah*Bl + Al*Bh). Block tile 64x64 (4 waves, 12 WMMA/K-step
// per wave). A-tiles staged with global_load_async_to_lds_b128 (ASYNCcnt)
// into double-buffered LDS; B-tiles converted once per block at staging.
// ---------------------------------------------------------------------------

typedef __attribute__((ext_vector_type(16))) __bf16 v16bf;
typedef __attribute__((ext_vector_type(4)))  __bf16 bf16x4;
typedef __attribute__((ext_vector_type(8)))  float  v8f;
typedef __attribute__((ext_vector_type(4)))  unsigned int u32x4;

union Frag16 { v16bf v; u32x4 q[2]; };

__device__ __forceinline__ void split_bf16(float x, __bf16& hi, __bf16& lo) {
    __bf16 h = (__bf16)x;            // RNE to bf16
    hi = h;
    lo = (__bf16)(x - (float)h);     // residual in bf16
}

enum { EPI_NONE = 0, EPI_BIAS = 1, EPI_BIAS_GELU = 2, EPI_BIAS_SOFTPLUS = 3,
       EPI_RESID = 4, EPI_BIAS_RESID = 5 };

// C[N x M] = A[N x K] (lda) @ Bw[K x M] (row-major). Block = 4 waves,
// block tile 64 x 64, K-step 32, double-buffered LDS, async A staging.
// Wave w computes rows 16w..16w+15 x all 64 cols (4 accumulators).
template <int EPI>
__global__ __launch_bounds__(128) void wmma_gemm_kernel(
    const float* __restrict__ A, int lda,
    const float* __restrict__ Bw,                 // K x M, ldb = M
    const float* __restrict__ bias,
    const float* __restrict__ resid, int ldr,
    float* __restrict__ C, int ldc,
    int K)
{
    __shared__ __align__(16) float  sA[2][64][36];     // fp32 A tile (144B rows)
    __shared__ __align__(16) __bf16 sBt_hi[2][64][40]; // B transposed [n][k]
    __shared__ __align__(16) __bf16 sBt_lo[2][64][40];

    const int tid  = threadIdx.x;
    const int lane = tid & 31;
    const int wave = tid >> 5;
    const int row0 = blockIdx.y * 64;
    const int col0 = blockIdx.x * 64;
    const int M    = gridDim.x * 64;          // == ldb

    const int mloc = (wave << 4) + (lane & 15);   // A row for this lane
    const int kbA  = (lane < 16) ? 0 : 8;         // A K-chunk base (ISA layout)
    const int kbB  = (lane < 16) ? 0 : 8;         // B K-chunk base (ISA layout)
    const int nloc = lane & 15;

    // ---- A tile: 4 x b128 async copies per thread (global -> LDS, ASYNCcnt)
    auto stage_a_async = [&](int k0, int buf) {
        #pragma unroll
        for (int i = 0; i < 4; ++i) {
            int f = i * 128 + tid;
            int r = f >> 3, c4 = (f & 7) << 2;
            unsigned lds_off =
                (unsigned)(size_t)&sA[buf][r][c4];      // low 32 bits = LDS addr
            unsigned long long gaddr =
                (unsigned long long)(size_t)&A[(size_t)(row0 + r) * lda + (k0 + c4)];
            asm volatile("global_load_async_to_lds_b128 %0, %1, off"
                         :: "v"(lds_off), "v"(gaddr) : "memory");
        }
    };

    // ---- B tile 32x64: 4 units/thread, unit = (kq, n): 4 consecutive k, 1 n.
    // Coalesced b32 loads; packed b64 LDS stores into transposed bf16 hi/lo.
    auto load_b = [&](int k0, float (&breg)[4][4]) {
        #pragma unroll
        for (int i = 0; i < 4; ++i) {
            int u = i * 128 + tid;
            int n = u & 63, kq = u >> 6;
            #pragma unroll
            for (int q = 0; q < 4; ++q)
                breg[i][q] = Bw[(size_t)(k0 + kq * 4 + q) * M + (col0 + n)];
        }
    };
    auto store_b = [&](const float (&breg)[4][4], int buf) {
        #pragma unroll
        for (int i = 0; i < 4; ++i) {
            int u = i * 128 + tid;
            int n = u & 63, kq = u >> 6;
            bf16x4 vh, vl;
            #pragma unroll
            for (int q = 0; q < 4; ++q) {
                __bf16 h, l; split_bf16(breg[i][q], h, l);
                vh[q] = h; vl[q] = l;
            }
            *(bf16x4*)&sBt_hi[buf][n][kq * 4] = vh;   // aligned b64 stores
            *(bf16x4*)&sBt_lo[buf][n][kq * 4] = vl;
        }
    };

    v8f acc[4] = {};

    // prologue: tile 0
    {
        float breg[4][4];
        stage_a_async(0, 0);
        load_b(0, breg);
        store_b(breg, 0);
    }
    asm volatile("s_wait_asynccnt 0x0" ::: "memory");
    __syncthreads();

    const int nk = K >> 5;
    for (int kt = 0; kt < nk; ++kt) {
        const int buf = kt & 1;
        const bool more = (kt + 1) < nk;
        float breg[4][4];
        if (more) {                       // overlap next-tile fetch with WMMA
            stage_a_async((kt + 1) << 5, buf ^ 1);
            load_b((kt + 1) << 5, breg);
        }

        // ---- gather A fragment (fp32 -> bf16 hi/lo), aligned b128 LDS loads
        const float4* pa0 = (const float4*)&sA[buf][mloc][kbA];
        const float4* pa1 = (const float4*)&sA[buf][mloc][kbA + 16];
        float4 a0 = pa0[0], a1 = pa0[1], a2 = pa1[0], a3 = pa1[1];
        float av[16] = {a0.x, a0.y, a0.z, a0.w, a1.x, a1.y, a1.z, a1.w,
                        a2.x, a2.y, a2.z, a2.w, a3.x, a3.y, a3.z, a3.w};
        Frag16 aH, aL;
        #pragma unroll
        for (int e = 0; e < 16; ++e) {
            __bf16 h, l; split_bf16(av[e], h, l);
            aH.v[e] = h; aL.v[e] = l;
        }

        // ---- 4 N-subtiles: 3 compensated WMMAs each (12 WMMA / K-step)
        #pragma unroll
        for (int j = 0; j < 4; ++j) {
            const int nrow = (j << 4) + nloc;
            Frag16 bH, bL;
            bH.q[0] = *(const u32x4*)&sBt_hi[buf][nrow][kbB];
            bH.q[1] = *(const u32x4*)&sBt_hi[buf][nrow][kbB + 16];
            bL.q[0] = *(const u32x4*)&sBt_lo[buf][nrow][kbB];
            bL.q[1] = *(const u32x4*)&sBt_lo[buf][nrow][kbB + 16];
            acc[j] = __builtin_amdgcn_wmma_f32_16x16x32_bf16(
                false, aH.v, false, bH.v, (short)0, acc[j], false, false);
            acc[j] = __builtin_amdgcn_wmma_f32_16x16x32_bf16(
                false, aH.v, false, bL.v, (short)0, acc[j], false, false);
            acc[j] = __builtin_amdgcn_wmma_f32_16x16x32_bf16(
                false, aL.v, false, bH.v, (short)0, acc[j], false, false);
        }

        if (more) {
            store_b(breg, buf ^ 1);
            asm volatile("s_wait_asynccnt 0x0" ::: "memory");
        }
        __syncthreads();
    }

    // ---- epilogue (C/D layout: VGPR r -> row r + (lane<16?0:8), col = lane&15)
    #pragma unroll
    for (int j = 0; j < 4; ++j) {
        const int colg = col0 + (j << 4) + nloc;
        float bval = 0.0f;
        if (EPI == EPI_BIAS || EPI == EPI_BIAS_GELU || EPI == EPI_BIAS_SOFTPLUS ||
            EPI == EPI_BIAS_RESID)
            bval = bias[colg];
        #pragma unroll
        for (int r = 0; r < 8; ++r) {
            int rowg = row0 + (wave << 4) + r + ((lane < 16) ? 0 : 8);
            float v = acc[j][r] + bval;
            if (EPI == EPI_BIAS_GELU)
                v = 0.5f * v * (1.0f + erff(v * 0.70710678118654752f));
            if (EPI == EPI_BIAS_SOFTPLUS)
                v = (v > 20.0f) ? v : log1pf(__expf(v));
            if (EPI == EPI_RESID || EPI == EPI_BIAS_RESID)
                v += resid[(size_t)rowg * ldr + colg];
            C[(size_t)rowg * ldc + colg] = v;
        }
    }
}

// LayerNorm over 512 cols, one wave per row; optional residual add after LN.
__global__ __launch_bounds__(256) void layernorm_kernel(
    const float* __restrict__ x, const float* __restrict__ g,
    const float* __restrict__ b, const float* __restrict__ resid,
    float* __restrict__ y)
{
    const int row  = blockIdx.x * 8 + (threadIdx.x >> 5);
    const int lane = threadIdx.x & 31;
    const float* xr = x + (size_t)row * 512;

    float vals[16];
    float s = 0.0f;
    #pragma unroll
    for (int i = 0; i < 16; ++i) { vals[i] = xr[lane + 32 * i]; s += vals[i]; }
    #pragma unroll
    for (int off = 16; off > 0; off >>= 1) s += __shfl_xor(s, off, 32);
    const float mean = s * (1.0f / 512.0f);

    float vs = 0.0f;
    #pragma unroll
    for (int i = 0; i < 16; ++i) { float d = vals[i] - mean; vs += d * d; }
    #pragma unroll
    for (int off = 16; off > 0; off >>= 1) vs += __shfl_xor(vs, off, 32);
    const float rstd = rsqrtf(vs * (1.0f / 512.0f) + 1e-5f);

    #pragma unroll
    for (int i = 0; i < 16; ++i) {
        int c = lane + 32 * i;
        float o = (vals[i] - mean) * rstd * g[c] + b[c];
        if (resid) o += resid[(size_t)row * 512 + c];
        y[(size_t)row * 512 + c] = o;
    }
}

// Causal depthwise conv (D_CONV=4) over xc = xz[:, :1024], then SiLU.
__global__ __launch_bounds__(256) void conv_silu_kernel(
    const float* __restrict__ xz, const float* __restrict__ w,
    const float* __restrict__ bias, float* __restrict__ u)
{
    int idx = blockIdx.x * blockDim.x + threadIdx.x;   // B*L*1024
    int d = idx & 1023;
    int t = (idx >> 10) & 2047;
    int b = idx >> 21;
    float acc = bias[d];
    #pragma unroll
    for (int j = 0; j < 4; ++j) {
        int tt = t - 3 + j;
        if (tt >= 0)
            acc += xz[((size_t)(b * 2048 + tt)) * 2048 + d] * w[d * 4 + j];
    }
    u[idx] = acc / (1.0f + __expf(-acc));   // silu
}

// Selective scan: one thread per (b, d) channel, h[16] in registers,
// B_t/C_t broadcast through LDS. Sequential over L=2048.
__global__ __launch_bounds__(256) void scan_kernel(
    const float* __restrict__ u, const float* __restrict__ delta,
    const float* __restrict__ xdbl, const float* __restrict__ A_log,
    const float* __restrict__ D_ssm, float* __restrict__ y)
{
    const int b = blockIdx.x >> 2;
    const int d = ((blockIdx.x & 3) << 8) + threadIdx.x;

    float Arow[16];
    #pragma unroll
    for (int s = 0; s < 16; ++s) Arow[s] = -__expf(A_log[d * 16 + s]);
    const float Dv = D_ssm[d];

    float h[16];
    #pragma unroll
    for (int s = 0; s < 16; ++s) h[s] = 0.0f;

    __shared__ float bc[32];   // [0..15]=B_t, [16..31]=C_t
    for (int t = 0; t < 2048; ++t) {
        const size_t rt = (size_t)b * 2048 + t;
        if (threadIdx.x < 32) bc[threadIdx.x] = xdbl[rt * 64 + 32 + threadIdx.x];
        __syncthreads();
        const float dlt = delta[rt * 1024 + d];
        const float uv  = u[rt * 1024 + d];
        const float du  = dlt * uv;
        float yv = 0.0f;
        #pragma unroll
        for (int s = 0; s < 16; ++s) {
            float dA = __expf(dlt * Arow[s]);
            h[s] = h[s] * dA + du * bc[s];
            yv  += h[s] * bc[16 + s];
        }
        y[rt * 1024 + d] = yv + uv * Dv;
        __syncthreads();
    }
}

// y2 = y * silu(z),  z = xz[:, 1024:2048]
__global__ __launch_bounds__(256) void gate_kernel(
    const float* __restrict__ y, const float* __restrict__ xz,
    float* __restrict__ y2)
{
    int idx = blockIdx.x * blockDim.x + threadIdx.x;   // B*L*1024
    int d = idx & 1023;
    int row = idx >> 10;
    float z = xz[(size_t)row * 2048 + 1024 + d];
    y2[idx] = y[idx] * (z / (1.0f + __expf(-z)));
}

// ---------------------------------------------------------------------------

extern "C" void kernel_launch(void* const* d_in, const int* in_sizes, int n_in,
                              void* d_out, int out_size, void* d_ws, size_t ws_size,
                              hipStream_t stream) {
    (void)in_sizes; (void)n_in; (void)out_size; (void)ws_size;

    const float* x_tokens   = (const float*)d_in[0];   // (4,2048,1024)
    const float* w_red      = (const float*)d_in[1];   // (1024,512)
    const float* w_exp      = (const float*)d_in[2];   // (512,1024)
    const float* ln1_g      = (const float*)d_in[3];
    const float* ln1_b      = (const float*)d_in[4];
    const float* ln2_g      = (const float*)d_in[5];
    const float* ln2_b      = (const float*)d_in[6];
    const float* in_proj_w  = (const float*)d_in[7];   // (512,2048)
    const float* conv_w     = (const float*)d_in[8];   // (1024,4)
    const float* conv_b     = (const float*)d_in[9];
    const float* x_proj_w   = (const float*)d_in[10];  // (1024,64)
    const float* dt_proj_w  = (const float*)d_in[11];  // (32,1024)
    const float* dt_proj_b  = (const float*)d_in[12];
    const float* A_log      = (const float*)d_in[13];  // (1024,16)
    const float* D_ssm      = (const float*)d_in[14];
    const float* out_proj_w = (const float*)d_in[15];  // (1024,512)
    const float* fc1_w      = (const float*)d_in[16];  // (512,2048)
    const float* fc1_b      = (const float*)d_in[17];
    const float* fc2_w      = (const float*)d_in[18];  // (2048,512)
    const float* fc2_b      = (const float*)d_in[19];
    float* out = (float*)d_out;

    // workspace layout (float offsets); reused buffers noted
    float* ws = (float*)d_ws;
    float* xflat = ws;                       //  8192x512
    float* xn1   = ws + 4194304;             //  8192x512
    float* xz    = ws + 8388608;             //  8192x2048
    float* u     = ws + 25165824;            //  8192x1024
    float* xdbl  = ws + 33554432;            //  8192x64
    float* delta = ws + 34078720;            //  8192x1024
    float* y     = ws + 42467328;            //  8192x1024
    float* xmamba= ws + 50855936;            //  8192x512
    float* y2        = delta;                // reuse (delta dead after scan)
    float* mlp_mid   = xz;                   // reuse (z dead after gate)
    float* mamba_out = xn1;                  // reuse (xn1 dead after in_proj)
    float* sum_out   = xflat;                // reuse (x_res dead after ln2)

    const dim3 blk(128);
    const int NY = 128;                      // 8192 rows / 64

    // 1) x_flat = x_tokens @ w_red
    wmma_gemm_kernel<EPI_NONE><<<dim3(8, NY), blk, 0, stream>>>(
        x_tokens, 1024, w_red, nullptr, nullptr, 0, xflat, 512, 1024);
    // 2) x_n1 = LN1(x_flat)
    layernorm_kernel<<<1024, 256, 0, stream>>>(xflat, ln1_g, ln1_b, nullptr, xn1);
    // 3) xz = x_n1 @ in_proj_w
    wmma_gemm_kernel<EPI_NONE><<<dim3(32, NY), blk, 0, stream>>>(
        xn1, 512, in_proj_w, nullptr, nullptr, 0, xz, 2048, 512);
    // 4) u = silu(causal_dwconv(xc))
    conv_silu_kernel<<<32768, 256, 0, stream>>>(xz, conv_w, conv_b, u);
    // 5) x_dbl = u @ x_proj_w
    wmma_gemm_kernel<EPI_NONE><<<dim3(1, NY), blk, 0, stream>>>(
        u, 1024, x_proj_w, nullptr, nullptr, 0, xdbl, 64, 1024);
    // 6) delta = softplus(dt @ dt_proj_w + dt_proj_b)   (dt = x_dbl[:, :32])
    wmma_gemm_kernel<EPI_BIAS_SOFTPLUS><<<dim3(16, NY), blk, 0, stream>>>(
        xdbl, 64, dt_proj_w, dt_proj_b, nullptr, 0, delta, 1024, 32);
    // 7) selective scan -> y (includes + u * D)
    scan_kernel<<<16, 256, 0, stream>>>(u, delta, xdbl, A_log, D_ssm, y);
    // 8) y2 = y * silu(z)
    gate_kernel<<<32768, 256, 0, stream>>>(y, xz, y2);
    // 9) x_mamba = y2 @ out_proj_w
    wmma_gemm_kernel<EPI_NONE><<<dim3(8, NY), blk, 0, stream>>>(
        y2, 1024, out_proj_w, nullptr, nullptr, 0, xmamba, 512, 1024);
    // 10) mamba_out = LN2(x_mamba) + x_res
    layernorm_kernel<<<1024, 256, 0, stream>>>(xmamba, ln2_g, ln2_b, xflat, mamba_out);
    // 11) mlp_mid = gelu(mamba_out @ fc1_w + fc1_b)
    wmma_gemm_kernel<EPI_BIAS_GELU><<<dim3(32, NY), blk, 0, stream>>>(
        mamba_out, 512, fc1_w, fc1_b, nullptr, 0, mlp_mid, 2048, 512);
    // 12) sum = (mlp_mid @ fc2_w + fc2_b) + mamba_out
    wmma_gemm_kernel<EPI_BIAS_RESID><<<dim3(8, NY), blk, 0, stream>>>(
        mlp_mid, 2048, fc2_w, fc2_b, mamba_out, 512, sum_out, 512, 2048);
    // 13) out = sum @ w_exp + x_tokens
    wmma_gemm_kernel<EPI_RESID><<<dim3(16, NY), blk, 0, stream>>>(
        sum_out, 512, w_exp, nullptr, x_tokens, 1024, out, 1024, 512);
}
